// MSDeformableAttention3D_14465449853379
// MI455X (gfx1250) — compile-verified, matrix operand order checked by reading, not run on hardware
//
#include <hip/hip_runtime.h>
#include <hip/hip_bf16.h>

// MS Deformable Attention 3D for MI455X (gfx1250, wave32, WMMA + TDM).
//
// Pipeline:
//   1) fp32 -> bf16 conversion of query/value/weights into workspace
//   2) WMMA GEMM (B tile staged in LDS via TDM tensor_load_to_lds):
//        vproj = value @ Wv.T + bv             [NV, 256]
//        qproj = query @ [Ws;Wa].T + [b_s;ba]  [NQ, 192]
//   3) softmax + bilinear-gather + weighted-sum kernel [NQ, 256]

typedef __attribute__((ext_vector_type(16))) __bf16 v16bf;
typedef __attribute__((ext_vector_type(8)))  __bf16 v8bf;
typedef __attribute__((ext_vector_type(8)))  float  v8f;

typedef __attribute__((ext_vector_type(4))) unsigned int v4u;
typedef __attribute__((ext_vector_type(8))) int          v8i;
typedef __attribute__((ext_vector_type(4))) int          v4i;

#define NQ_    40000
#define H_     100
#define W_     148
#define NV_    (H_ * W_)
#define C_     256
#define HEADS_ 8
#define NPTS_  8
#define D_     32
#define K_     256

// LDS B-tile row stride in bf16 elements: 256 data + 8 pad (16B) so that
// consecutive rows shift by 4 LDS banks -> conflict-free 16-lane b128 reads.
#define LDSROW_ 264

// ---------------------------------------------------------------------------
// fp32 -> bf16 elementwise conversion
// ---------------------------------------------------------------------------
__global__ void cvt_f32_bf16(const float* __restrict__ in,
                             __bf16* __restrict__ out, int n) {
  int i = blockIdx.x * blockDim.x + threadIdx.x;
  if (i < n) out[i] = (__bf16)in[i];
}

// ---------------------------------------------------------------------------
// TDM: load a 64-row x 256-col bf16 tile (row stride 256) from global into
// LDS, inserting 4 DWORDs of padding after every 128 DWORDs (i.e. 16B pad
// per 512B row -> LDS row stride 528B = LDSROW_ bf16).
// Descriptor layout per CDNA5 ISA 8.3/8.4 (groups 2/3 zero => 2-D tensor).
// ---------------------------------------------------------------------------
__device__ __forceinline__ void tdm_load_tile64x256(unsigned lds_off,
                                                    unsigned long long gaddr) {
  v4u g0 = {
      1u,                                             // count=1, user desc
      lds_off,                                        // lds_addr (bytes)
      (unsigned)(gaddr & 0xFFFFFFFFu),                // global_addr[31:0]
      (unsigned)(((gaddr >> 32) & 0x01FFFFFFu)        // global_addr[56:32]
                 | 0x80000000u)                       // type=2 ("image")
  };
  v8i g1 = {
      (int)((1u << 16)        // data_size = 1 -> 2 bytes/elem
            | (1u << 20)      // pad_enable
            | (6u << 22)      // pad_interval: 128 DWORDs (one 512B row)
            | (3u << 25)),    // pad_amount: 4 DWORDs (16B)
      (int)(256u << 16),      // tensor_dim0[15:0] = 256 (elements per row)
      (int)(64u << 16),       // tensor_dim1[15:0] = 64 rows
      (int)(256u << 16),      // tile_dim0 = 256
      (int)64,                // tile_dim1 = 64, tile_dim2 = 0
      (int)256,               // tensor_dim0_stride[31:0] = 256 elements
      0, 0                    // stride hi / dim1_stride (unused, 2-D tile)
  };
  v4i gz = {0, 0, 0, 0};
#if __clang_major__ >= 23
  v8i gz8 = {0, 0, 0, 0, 0, 0, 0, 0};
  __builtin_amdgcn_tensor_load_to_lds(g0, g1, gz, gz, gz8, 0);
#else
  __builtin_amdgcn_tensor_load_to_lds(g0, g1, gz, gz, 0);
#endif
}

// ---------------------------------------------------------------------------
// WMMA bf16 GEMM with bias, D = A @ B.T + bias
//   A:  [M, 256] bf16 row-major
//   B0: rows [0, N0) of B.T as [N0, 256] bf16 (weight rows), B1: rows [N0, N)
//   D:  [M, N] fp32 row-major
// Block = 8 waves. The block's 64x256 B tile is TDM-staged into LDS once;
// each wave computes a 16x64 strip (4 accumulators, A fragment reused 4x).
// grid = (ceil(M/128), N/64); N0 and 64-column groups are 64-aligned so each
// tile comes entirely from B0 or B1.
// ---------------------------------------------------------------------------
__device__ __forceinline__ v8bf ld8(const __bf16* p) {
  return *(const v8bf*)p;
}
__device__ __forceinline__ v16bf cat16(v8bf lo, v8bf hi) {
  return __builtin_shufflevector(lo, hi, 0, 1, 2, 3, 4, 5, 6, 7,
                                 8, 9, 10, 11, 12, 13, 14, 15);
}

__global__ void gemm_wmma_bias(const __bf16* __restrict__ A, int M,
                               const __bf16* __restrict__ B0, int N0,
                               const float* __restrict__ bias0,
                               const __bf16* __restrict__ B1,
                               const float* __restrict__ bias1,
                               float* __restrict__ D, int N) {
  __shared__ __bf16 Bs[64 * LDSROW_];

  const int wave = threadIdx.x >> 5;
  const int lane = threadIdx.x & 31;
  const int col_base = blockIdx.y * 64;

  // Stage B tile: one TDM descriptor issued by wave 0, tracked by TENSORcnt.
  if (wave == 0) {
    const __bf16* src = (col_base < N0)
                            ? (B0 + (size_t)col_base * K_)
                            : (B1 + (size_t)(col_base - N0) * K_);
    tdm_load_tile64x256((unsigned)(unsigned long long)Bs,
                        (unsigned long long)src);
    __builtin_amdgcn_s_wait_tensorcnt(0);
  }
  __syncthreads();

  const int row_tile = blockIdx.x * 8 + wave;
  if (row_tile * 16 >= M) return;  // wave-uniform exit, after the barrier

  const int r    = lane & 15;  // M (A) / N (B) index within 16x16 tile
  const int half = lane >> 4;  // K sub-chunk select per ISA 16-bit layout
  const int kb   = half * 8;

  const __bf16* Arow = A + (size_t)(row_tile * 16 + r) * K_;

  v8f acc[4];
#pragma unroll
  for (int c = 0; c < 4; ++c) {
    int n = col_base + c * 16 + r;
    float bz = (n < N0) ? bias0[n] : bias1[n - N0];
    acc[c] = (v8f){bz, bz, bz, bz, bz, bz, bz, bz};
  }

#pragma unroll
  for (int k0 = 0; k0 < K_; k0 += 32) {
    // A 16x32 bf16 fragment (ISA 7.12.2): lane holds K = [k0+kb, k0+kb+8)
    // and [k0+16+kb, k0+16+kb+8) of row (lane&15).
    v16bf a = cat16(ld8(Arow + k0 + kb), ld8(Arow + k0 + 16 + kb));
#pragma unroll
    for (int c = 0; c < 4; ++c) {
      const __bf16* bp = Bs + (size_t)(c * 16 + r) * LDSROW_ + k0 + kb;
      v16bf b = cat16(ld8(bp), ld8(bp + 16));
      acc[c] = __builtin_amdgcn_wmma_f32_16x16x32_bf16(
          false, a, false, b, (short)0, acc[c], false, false);
    }
  }

  // D layout (ISA 7.12.2): VGPR i of lane L -> row = i + 8*(L>>4), col = L&15
#pragma unroll
  for (int c = 0; c < 4; ++c) {
    int col = col_base + c * 16 + r;
#pragma unroll
    for (int i = 0; i < 8; ++i) {
      int row = row_tile * 16 + i + 8 * half;
      D[(size_t)row * N + col] = acc[c][i];
    }
  }
}

// ---------------------------------------------------------------------------
// Softmax + bilinear sampling + weighted sum.
// One block per query, one wave per head, one lane per channel (D=32=wave32).
//   v:    [NV, 256] fp32  (= [NV][HEADS][32])
//   qp:   [NQ, 192] fp32  (cols 0..127 offsets = [H][8][2], 128..191 logits)
//   refp: [NQ, 4, 2] fp32
//   out:  [NQ, 256] fp32
// ---------------------------------------------------------------------------
__global__ void msda_sample(const float* __restrict__ v,
                            const float* __restrict__ qp,
                            const float* __restrict__ refp,
                            float* __restrict__ out) {
  const int q    = blockIdx.x;
  const int h    = threadIdx.x >> 5;
  const int lane = threadIdx.x & 31;

  const float* base = qp + (size_t)q * 192;
  const float* lg   = base + 128 + h * NPTS_;

  // softmax over the 8 sampling-point logits (uniform across lanes)
  float m = lg[0];
#pragma unroll
  for (int p = 1; p < NPTS_; ++p) m = fmaxf(m, lg[p]);
  float e[NPTS_];
  float s = 0.f;
#pragma unroll
  for (int p = 0; p < NPTS_; ++p) {
    e[p] = __expf(lg[p] - m);
    s += e[p];
  }
  const float inv = __frcp_rn(s);

  const float* o  = base + h * (NPTS_ * 2);
  const float* rp = refp + (size_t)q * 8;  // [4][2]

  const int ch = h * D_ + lane;
  float acc = 0.f;

#pragma unroll
  for (int p = 0; p < NPTS_; ++p) {
    // loc = ref[p % 4] + off[p] / [W, H]   (normalized [0,1] coords)
    float lx = rp[(p & 3) * 2 + 0] + o[p * 2 + 0] * (1.0f / (float)W_);
    float ly = rp[(p & 3) * 2 + 1] + o[p * 2 + 1] * (1.0f / (float)H_);
    // align_corners=False unnormalization
    float x = lx * (float)W_ - 0.5f;
    float y = ly * (float)H_ - 0.5f;
    float x0f = floorf(x), y0f = floorf(y);
    int x0 = (int)x0f, y0 = (int)y0f;
    float wx = x - x0f, wy = y - y0f;

    auto g = [&](int yy, int xx) -> float {
      if (yy < 0 || yy >= H_ || xx < 0 || xx >= W_) return 0.f;
      return v[(size_t)(yy * W_ + xx) * C_ + ch];  // coalesced across lanes
    };
    float v00 = g(y0, x0);
    float v01 = g(y0, x0 + 1);
    float v10 = g(y0 + 1, x0);
    float v11 = g(y0 + 1, x0 + 1);

    float bil = (1.f - wy) * ((1.f - wx) * v00 + wx * v01) +
                wy * ((1.f - wx) * v10 + wx * v11);
    acc = fmaf(e[p] * inv, bil, acc);
  }

  out[(size_t)q * C_ + ch] = acc;
}

// ---------------------------------------------------------------------------
// Host launcher
// ---------------------------------------------------------------------------
extern "C" void kernel_launch(void* const* d_in, const int* in_sizes, int n_in,
                              void* d_out, int out_size, void* d_ws,
                              size_t ws_size, hipStream_t stream) {
  const float* query = (const float*)d_in[0];  // [1, NQ, 256]
  const float* value = (const float*)d_in[1];  // [1, NV, 256]
  const float* refp  = (const float*)d_in[2];  // [1, NQ, 4, 2]
  // d_in[3] = spatial_shapes (constants hardcoded: H=100, W=148)
  const float* Wv  = (const float*)d_in[4];    // [256, 256]
  const float* bv  = (const float*)d_in[5];    // [256]
  const float* Ws  = (const float*)d_in[6];    // [128, 256]
  const float* b_s = (const float*)d_in[7];    // [128]
  const float* Wa  = (const float*)d_in[8];    // [64, 256]
  const float* ba  = (const float*)d_in[9];    // [64]
  float* out = (float*)d_out;

  // Workspace carve-out (256B aligned); ~74 MB total.
  char* ws = (char*)d_ws;
  size_t cur = 0;
  auto carve = [&](size_t bytes) -> char* {
    char* p = ws + cur;
    cur = (cur + bytes + 255) & ~(size_t)255;
    return p;
  };
  __bf16* q_bf   = (__bf16*)carve((size_t)NQ_ * C_ * 2);
  __bf16* v_bf   = (__bf16*)carve((size_t)NV_ * C_ * 2);
  __bf16* Wv_bf  = (__bf16*)carve((size_t)C_ * C_ * 2);
  __bf16* Ws_bf  = (__bf16*)carve((size_t)128 * C_ * 2);
  __bf16* Wa_bf  = (__bf16*)carve((size_t)64 * C_ * 2);
  float*  vproj  = (float*)carve((size_t)NV_ * C_ * 4);
  float*  qproj  = (float*)carve((size_t)NQ_ * 192 * 4);

  auto cvt = [&](const float* src, __bf16* dst, int n) {
    cvt_f32_bf16<<<(n + 255) / 256, 256, 0, stream>>>(src, dst, n);
  };
  cvt(query, q_bf, NQ_ * C_);
  cvt(value, v_bf, NV_ * C_);
  cvt(Wv, Wv_bf, C_ * C_);
  cvt(Ws, Ws_bf, 128 * C_);
  cvt(Wa, Wa_bf, 64 * C_);

  // vproj = value @ Wv.T + bv : M=14800, N=256
  {
    dim3 grid((NV_ / 16 + 7) / 8, C_ / 64);
    gemm_wmma_bias<<<grid, 256, 0, stream>>>(v_bf, NV_, Wv_bf, C_, bv, Wv_bf,
                                             bv, vproj, C_);
  }
  // qproj = query @ [Ws;Wa].T + [b_s;ba] : M=40000, N=192
  {
    dim3 grid((NQ_ / 16 + 7) / 8, 192 / 64);
    gemm_wmma_bias<<<grid, 256, 0, stream>>>(q_bf, NQ_, Ws_bf, 128, b_s, Wa_bf,
                                             ba, qproj, 192);
  }
  // sampling: one block per query, one wave per head
  msda_sample<<<NQ_, 256, 0, stream>>>(vproj, qproj, refp, out);
}
